// MambaMIDIGenerator_8555574854448
// MI455X (gfx1250) — matrix operand order repaired and checked by reading, not compiled, not run on hardware
//
#include <hip/hip_runtime.h>
#include <stdint.h>
#include <stddef.h>

// ---- model constants (must match reference) ----
#define D_MODEL 768
#define DI      1536          // D_INNER
#define DS      16            // D_STATE
#define NL      8
#define NVOCAB  512
#define LSEQ    2048
#define NB      8
#define MROWS   (NB * LSEQ)   // 16384 token rows

typedef __bf16 bf16x16 __attribute__((ext_vector_type(16)));
typedef float  f32x8   __attribute__((ext_vector_type(8)));
typedef unsigned u32x4 __attribute__((ext_vector_type(4)));
typedef int      i32x8 __attribute__((ext_vector_type(8)));
typedef int      i32x4 __attribute__((ext_vector_type(4)));

#if defined(__gfx1250__) && __has_builtin(__builtin_amdgcn_tensor_load_to_lds)
#define HAVE_TDM 1
#else
#define HAVE_TDM 0
#endif

struct __align__(16) U4 { unsigned a, b, c, d; };
struct __align__(16) F4 { float x, y, z, w; };

__device__ __forceinline__ unsigned short f2bf(float f) {
  unsigned u = __float_as_uint(f);
  u += 0x7FFFu + ((u >> 16) & 1u);          // round-to-nearest-even
  return (unsigned short)(u >> 16);
}
__device__ __forceinline__ float bf2f(unsigned short h) {
  return __uint_as_float(((unsigned)h) << 16);
}
__device__ __forceinline__ float dev_silu(float v) { return v / (1.f + __expf(-v)); }

// ------------------------------------------------------------------
// f32 -> bf16 bulk convert (weights)
// ------------------------------------------------------------------
__global__ void k_f32_to_bf16(const float* __restrict__ in,
                              unsigned short* __restrict__ out, size_t n) {
  size_t i = (size_t)blockIdx.x * blockDim.x + threadIdx.x;
  if (i < n) out[i] = f2bf(in[i]);
}

// ------------------------------------------------------------------
// conditioning vector: cond[b,d] = cond_b[d] + sum_i pre[b,i]*cond_w[d,i]
// ------------------------------------------------------------------
__global__ void k_cond(const int* __restrict__ emotion, const int* __restrict__ style,
                       const int* __restrict__ keysig, const int* __restrict__ modei,
                       const float* __restrict__ bpm,
                       const float* __restrict__ emo_emb, const float* __restrict__ sty_emb,
                       const float* __restrict__ key_emb, const float* __restrict__ mode_emb,
                       const float* __restrict__ bpm_w, const float* __restrict__ bpm_b,
                       const float* __restrict__ cond_w, const float* __restrict__ cond_b,
                       float* __restrict__ cond_out) {
  int t = blockIdx.x * blockDim.x + threadIdx.x;
  if (t >= NB * D_MODEL) return;
  int bi = t / D_MODEL, d = t % D_MODEL;
  const float* e  = emo_emb  + emotion[bi] * D_MODEL;
  const float* s  = sty_emb  + style[bi]   * D_MODEL;
  const float* k  = key_emb  + keysig[bi]  * D_MODEL;
  const float* mo = mode_emb + modei[bi]   * D_MODEL;
  float bp = bpm[bi];
  const float* w = cond_w + (size_t)d * (5 * D_MODEL);
  float acc = cond_b[d];
  for (int i = 0; i < D_MODEL; ++i) {
    acc += e[i]  * w[i];
    acc += s[i]  * w[D_MODEL + i];
    acc += k[i]  * w[2 * D_MODEL + i];
    acc += mo[i] * w[3 * D_MODEL + i];
    acc += (bp * bpm_w[i] + bpm_b[i]) * w[4 * D_MODEL + i];
  }
  cond_out[t] = acc;
}

// ------------------------------------------------------------------
// x[b,l,d] = tok_emb[tokens[b,l],d] + cond[b,d] + pos_emb[l,d]
// ------------------------------------------------------------------
__global__ void k_embed(const int* __restrict__ tokens, const float* __restrict__ tok_emb,
                        const float* __restrict__ cond, const float* __restrict__ pos_emb,
                        float* __restrict__ x) {
  size_t i = (size_t)blockIdx.x * blockDim.x + threadIdx.x;
  if (i >= (size_t)MROWS * D_MODEL) return;
  int d = (int)(i % D_MODEL);
  int m = (int)(i / D_MODEL);
  int l = m & (LSEQ - 1);
  int b = m >> 11;                      // /2048
  int tok = tokens[m];
  x[i] = tok_emb[(size_t)tok * D_MODEL + d] + cond[b * D_MODEL + d]
       + pos_emb[(size_t)l * D_MODEL + d];
}

// ------------------------------------------------------------------
// LayerNorm over D_MODEL=768, one wave32 per row, bf16 output
// ------------------------------------------------------------------
__global__ __launch_bounds__(256)
void k_layernorm(const float* __restrict__ x, const float* __restrict__ g,
                 const float* __restrict__ b, unsigned short* __restrict__ out, int rows) {
  int wv   = (blockIdx.x * blockDim.x + threadIdx.x) >> 5;
  int lane = threadIdx.x & 31;
  if (wv >= rows) return;
  const float* xr = x + (size_t)wv * D_MODEL;
  float s = 0.f, s2 = 0.f;
  for (int j = lane; j < D_MODEL; j += 32) { float v = xr[j]; s += v; s2 += v * v; }
  for (int off = 16; off; off >>= 1) {
    s  += __shfl_xor(s,  off, 32);
    s2 += __shfl_xor(s2, off, 32);
  }
  float mu  = s * (1.f / D_MODEL);
  float var = s2 * (1.f / D_MODEL) - mu * mu;
  float inv = rsqrtf(var + 1e-5f);
  unsigned short* orow = out + (size_t)wv * D_MODEL;
  for (int j = lane; j < D_MODEL; j += 32)
    orow[j] = f2bf((xr[j] - mu) * inv * g[j] + b[j]);
}

// ------------------------------------------------------------------
// TDM: DMA a [tileRows x tileK] bf16 tile (row stride rowStrideElems) from
// global into LDS at byte offset ldsByteAddr, packed row-major contiguously.
// D# bit layout per CDNA5 ISA ch.8 (group0: count/lds_addr/global_addr/type,
// group1: data_size/tensor dims/tile dims/stride, groups 2-3 zero -> 2-D).
// This toolchain's builtin takes 6 args (g0 u32x4, g1 i32x8, g2 i32x4,
// g3 i32x4, extra i32x8, cpol i32).
// ------------------------------------------------------------------
#if HAVE_TDM
__device__ __forceinline__ void tdm_load_tile_2d(unsigned ldsByteAddr,
                                                 const unsigned short* gtile,
                                                 unsigned tileK, unsigned tileRows,
                                                 unsigned rowStrideElems) {
  unsigned long long ga = (unsigned long long)(uintptr_t)gtile;
  u32x4 g0;
  g0[0] = 1u;                                            // count=1, user desc
  g0[1] = ldsByteAddr;                                   // lds_addr (bytes)
  g0[2] = (unsigned)ga;                                  // global_addr[31:0]
  g0[3] = (unsigned)((ga >> 32) & 0x01FFFFFFull) | (2u << 30);  // addr[56:32] | type=2
  i32x8 g1;
  g1[0] = (int)(1u << 16);                               // data_size=1 (2 bytes)
  g1[1] = (int)((tileK & 0xFFFFu) << 16);                // tensor_dim0[15:0]
  g1[2] = (int)((tileK >> 16) & 0xFFFFu) |
          (int)((tileRows & 0xFFFFu) << 16);             // td0[31:16] | td1[15:0]
  g1[3] = (int)((tileRows >> 16) & 0xFFFFu) |
          (int)((tileK & 0xFFFFu) << 16);                // td1[31:16] | tile_dim0
  g1[4] = (int)(tileRows & 0xFFFFu);                     // tile_dim1 (tile_dim2=0)
  g1[5] = (int)rowStrideElems;                           // tensor_dim0_stride[31:0]
  g1[6] = 0;                                             // stride hi | td1_stride lo
  g1[7] = 0;
  i32x4 gz4 = {0, 0, 0, 0};
  i32x8 gz8 = {0, 0, 0, 0, 0, 0, 0, 0};
  __builtin_amdgcn_tensor_load_to_lds(g0, g1, gz4, gz4, gz8, 0);
}
#endif

// ------------------------------------------------------------------
// one 32-deep K step of WMMA compute from staged LDS tiles
// ------------------------------------------------------------------
union FragBF { U4 q[2]; bf16x16 v; };

__device__ __forceinline__ void gemm_step(const unsigned short* sA, const unsigned short* sB,
                                          int wm, int wn, int lm, int laneHalf,
                                          f32x8 acc[2][4]) {
  // A fragments: 16-bit A 16x32 layout — halves 0..7: K = 8*laneHalf+0..7,
  // halves 8..15: K = 16 + 8*laneHalf + 0..7, row = lane&15.
  FragBF fa0, fa1;
  {
    int r0 = (wm * 32 + lm) * 32 + laneHalf * 8;
    int r1 = (wm * 32 + 16 + lm) * 32 + laneHalf * 8;
    fa0.q[0] = *(const U4*)(sA + r0);
    fa0.q[1] = *(const U4*)(sA + r0 + 16);
    fa1.q[0] = *(const U4*)(sA + r1);
    fa1.q[1] = *(const U4*)(sA + r1 + 16);
  }
#pragma unroll
  for (int tn = 0; tn < 4; ++tn) {
    // B fragment: col = lane&15, K = 8*laneHalf + 0..7, 16 + 8*laneHalf + 0..7 (B(k,n) = W[n,k])
    FragBF fb;
    int br = (wn * 64 + tn * 16 + lm) * 32 + laneHalf * 8;
    fb.q[0] = *(const U4*)(sB + br);
    fb.q[1] = *(const U4*)(sB + br + 16);
    acc[0][tn] = __builtin_amdgcn_wmma_f32_16x16x32_bf16(
        false, fa0.v, false, fb.v, (short)0, acc[0][tn], false, false);
    acc[1][tn] = __builtin_amdgcn_wmma_f32_16x16x32_bf16(
        false, fa1.v, false, fb.v, (short)0, acc[1][tn], false, false);
  }
}

// ------------------------------------------------------------------
// WMMA bf16 GEMM: C[M,N] = A[M,K] @ W[N,K]^T
//   128x128 block tile, 8 waves (4x2), each wave 2x4 16x16 tiles, K step 32.
//   TDM path: double-buffered LDS, tiles DMA'd by the Tensor Data Mover and
//   overlapped with WMMA compute (TENSORcnt + barrier sync).
//   MODE 0: bf16 store; MODE 1: f32 in-place accumulate; MODE 2: f32 + bias.
// ------------------------------------------------------------------
template <int MODE>
__global__ __launch_bounds__(256)
void gemm_bf16_wmma(const unsigned short* __restrict__ A,
                    const unsigned short* __restrict__ W,
                    void* __restrict__ Out, const float* __restrict__ bias,
                    int N, int K) {
#if HAVE_TDM
  __shared__ unsigned short lA[2][128 * 32];
  __shared__ unsigned short lB[2][128 * 32];
#else
  __shared__ unsigned short lA[1][128 * 32];
  __shared__ unsigned short lB[1][128 * 32];
#endif
  const int tid      = threadIdx.x;
  const int lane     = tid & 31;
  const int wv       = tid >> 5;
  const int wm       = wv & 3;          // wave m-quadrant (0..3), 32 rows each
  const int wn       = wv >> 2;         // wave n-half (0..1), 64 cols each
  const int laneHalf = lane >> 4;       // 0 or 1
  const int lm       = lane & 15;
  const int mt0      = blockIdx.y * 128;
  const int nt0      = blockIdx.x * 128;

  f32x8 acc[2][4] = {};

#if HAVE_TDM
  const unsigned ldsA[2] = { (unsigned)(uintptr_t)&lA[0][0], (unsigned)(uintptr_t)&lA[1][0] };
  const unsigned ldsB[2] = { (unsigned)(uintptr_t)&lB[0][0], (unsigned)(uintptr_t)&lB[1][0] };
  const int nsteps = K / 32;
  if (wv == 0) {
    tdm_load_tile_2d(ldsA[0], A + (size_t)mt0 * K, 32u, 128u, (unsigned)K);
    tdm_load_tile_2d(ldsB[0], W + (size_t)nt0 * K, 32u, 128u, (unsigned)K);
    __builtin_amdgcn_s_wait_tensorcnt(0);
  }
  __syncthreads();
  for (int s = 0; s < nsteps; ++s) {
    const int cur = s & 1;
    if (wv == 0 && s + 1 < nsteps) {       // prefetch next tiles into other buffer
      int k0n = (s + 1) * 32;
      tdm_load_tile_2d(ldsA[cur ^ 1], A + (size_t)mt0 * K + k0n, 32u, 128u, (unsigned)K);
      tdm_load_tile_2d(ldsB[cur ^ 1], W + (size_t)nt0 * K + k0n, 32u, 128u, (unsigned)K);
    }
    gemm_step(lA[cur], lB[cur], wm, wn, lm, laneHalf, acc);
    if (wv == 0 && s + 1 < nsteps) __builtin_amdgcn_s_wait_tensorcnt(0);
    __syncthreads();
  }
#else
  const int row  = tid >> 1;            // 0..127 (LDS staging row)
  const int half = (tid & 1) * 16;      // 0 or 16 elements within the 32-wide K slab
  for (int k0 = 0; k0 < K; k0 += 32) {
    const U4* ga = (const U4*)(A + (size_t)(mt0 + row) * K + k0 + half);
    const U4* gb = (const U4*)(W + (size_t)(nt0 + row) * K + k0 + half);
    U4 a0 = ga[0], a1 = ga[1];
    U4 b0 = gb[0], b1 = gb[1];
    __syncthreads();                    // prior compute done before overwrite
    *(U4*)(&lA[0][row * 32 + half])     = a0;
    *(U4*)(&lA[0][row * 32 + half + 8]) = a1;
    *(U4*)(&lB[0][row * 32 + half])     = b0;
    *(U4*)(&lB[0][row * 32 + half + 8]) = b1;
    __syncthreads();
    gemm_step(lA[0], lB[0], wm, wn, lm, laneHalf, acc);
  }
#endif

  // C/D layout: lanes 0-15: M = r, N = lane; lanes 16-31: M = r+8, N = lane-16
#pragma unroll
  for (int tm = 0; tm < 2; ++tm) {
#pragma unroll
    for (int tn = 0; tn < 4; ++tn) {
      int mbase = mt0 + wm * 32 + tm * 16 + laneHalf * 8;
      int ncol  = nt0 + wn * 64 + tn * 16 + lm;
#pragma unroll
      for (int r = 0; r < 8; ++r) {
        size_t idx = (size_t)(mbase + r) * N + ncol;
        float v = acc[tm][tn][r];
        if (MODE == 0)       ((unsigned short*)Out)[idx] = f2bf(v);
        else if (MODE == 1)  ((float*)Out)[idx] += v;           // residual add
        else                 ((float*)Out)[idx] = v + bias[ncol];
      }
    }
  }
}

// ------------------------------------------------------------------
// causal depthwise conv (width 4) + SiLU over the xc half of xz (bf16)
// out[l] = silu( sum_{dl=0..3} w[3-dl] * xc[l-dl] )
// ------------------------------------------------------------------
__global__ void k_conv_silu(const unsigned short* __restrict__ xz,
                            const float* __restrict__ convw,   // layer slice [DI,4]
                            unsigned short* __restrict__ xcs) {
  size_t i = (size_t)blockIdx.x * blockDim.x + threadIdx.x;
  if (i >= (size_t)MROWS * DI) return;
  int c = (int)(i % DI);
  int m = (int)(i / DI);
  int l = m & (LSEQ - 1);
  const float* w = convw + (size_t)c * 4;
  float acc = 0.f;
#pragma unroll
  for (int dl = 0; dl < 4; ++dl)
    if (l >= dl) acc += w[3 - dl] * bf2f(xz[(size_t)(m - dl) * (2 * DI) + c]);
  xcs[i] = f2bf(dev_silu(acc));
}

// ------------------------------------------------------------------
// xproj: dtb[m, 0:32] = xcs[m, :] @ xproj_w^T   (xproj_w [32, DI], f32)
// One wave per m-row (lane = output index), weights staged in LDS chunks.
// ------------------------------------------------------------------
#define XP_CHUNK 256
__global__ __launch_bounds__(256)
void k_xproj(const unsigned short* __restrict__ xcs,
             const float* __restrict__ xw,        // layer slice [32, DI]
             float* __restrict__ dtb) {
  __shared__ float lw[32 * (XP_CHUNK + 1)];       // pad to avoid bank conflicts
  int lane = threadIdx.x & 31;
  int m    = blockIdx.x * 8 + (threadIdx.x >> 5);
  float acc = 0.f;
  for (int k0 = 0; k0 < DI; k0 += XP_CHUNK) {
    __syncthreads();
    for (int i = threadIdx.x; i < 32 * XP_CHUNK; i += 256) {
      int r = i >> 8, kk = i & (XP_CHUNK - 1);
      lw[r * (XP_CHUNK + 1) + kk] = xw[(size_t)r * DI + k0 + kk];
    }
    __syncthreads();
    const unsigned short* ar = xcs + (size_t)m * DI + k0;
    const float* wr = lw + lane * (XP_CHUNK + 1);
#pragma unroll 4
    for (int kk = 0; kk < XP_CHUNK; ++kk) acc += wr[kk] * bf2f(ar[kk]);
  }
  dtb[(size_t)m * 32 + lane] = acc;
}

// ------------------------------------------------------------------
// Fused: delta/bp projection + softplus/clip + sequential scan + D-residual
//        + SiLU(z) gating. One thread per (batch, channel). Writes bf16 y.
// ------------------------------------------------------------------
__global__ __launch_bounds__(256)
void k_scan_gate(const float* __restrict__ dtb,       // [MROWS, 32]
                 const float* __restrict__ dtw,       // layer [DI, DS]
                 const float* __restrict__ dtbias,    // layer [DI]
                 const float* __restrict__ bpw,       // layer [DI, DS]
                 const float* __restrict__ dparam,    // layer [DI]
                 const unsigned short* __restrict__ xcs,  // [MROWS, DI] bf16 (u)
                 const unsigned short* __restrict__ xz,   // [MROWS, 2*DI] bf16 (z half)
                 unsigned short* __restrict__ ybf) {      // [MROWS, DI] bf16
  int t = blockIdx.x * blockDim.x + threadIdx.x;       // NB*DI = 12288
  if (t >= NB * DI) return;
  int b = t / DI, c = t % DI;
  float wdt[DS], wbp[DS];
#pragma unroll
  for (int j = 0; j < DS; ++j) {
    wdt[j] = dtw[(size_t)c * DS + j];
    wbp[j] = bpw[(size_t)c * DS + j];
  }
  float bias = dtbias[c], dp = dparam[c];
  float h = 0.f;
  size_t mb = (size_t)b * LSEQ;
  for (int l = 0; l < LSEQ; ++l) {
    size_t m = mb + l;
    const F4* q = (const F4*)(dtb + m * 32);
    float qv[32];
#pragma unroll
    for (int i = 0; i < 8; ++i) {
      F4 tq = q[i];
      qv[4 * i] = tq.x; qv[4 * i + 1] = tq.y; qv[4 * i + 2] = tq.z; qv[4 * i + 3] = tq.w;
    }
    float dt = bias, bb = 0.f;
#pragma unroll
    for (int j = 0; j < DS; ++j) { dt += wdt[j] * qv[j]; bb += wbp[j] * qv[DS + j]; }
    float sp = (dt > 20.f) ? dt : log1pf(__expf(dt));   // softplus
    float delta = fminf(fmaxf(sp, 1e-4f), 1.0f);
    float u = bf2f(xcs[m * DI + c]);
    h = h * (1.f - delta) + u * bb;
    float z = bf2f(xz[m * (2 * DI) + DI + c]);
    ybf[m * DI + c] = f2bf((h + dp * u) * dev_silu(z));
  }
}

// ------------------------------------------------------------------
// host-side orchestration
// ------------------------------------------------------------------
extern "C" void kernel_launch(void* const* d_in, const int* in_sizes, int n_in,
                              void* d_out, int out_size, void* d_ws, size_t ws_size,
                              hipStream_t stream) {
  const int*   tokens   = (const int*)  d_in[0];
  const int*   emotion  = (const int*)  d_in[1];
  const int*   style    = (const int*)  d_in[2];
  const int*   keysig   = (const int*)  d_in[3];
  const int*   modei    = (const int*)  d_in[4];
  const float* bpm      = (const float*)d_in[5];
  const float* tok_emb  = (const float*)d_in[6];
  const float* emo_emb  = (const float*)d_in[7];
  const float* sty_emb  = (const float*)d_in[8];
  const float* key_emb  = (const float*)d_in[9];
  const float* mode_emb = (const float*)d_in[10];
  const float* bpm_w    = (const float*)d_in[11];
  const float* bpm_b    = (const float*)d_in[12];
  const float* cond_w   = (const float*)d_in[13];
  const float* cond_b   = (const float*)d_in[14];
  const float* pos_emb  = (const float*)d_in[15];
  const float* ln_g     = (const float*)d_in[16];
  const float* ln_b     = (const float*)d_in[17];
  const float* in_w     = (const float*)d_in[18];
  const float* conv_w   = (const float*)d_in[19];
  const float* xproj_w  = (const float*)d_in[20];
  const float* dt_w     = (const float*)d_in[21];
  const float* dt_b     = (const float*)d_in[22];
  const float* bproj_w  = (const float*)d_in[23];
  const float* d_param  = (const float*)d_in[24];
  const float* out_w    = (const float*)d_in[25];
  const float* final_g  = (const float*)d_in[26];
  const float* final_b  = (const float*)d_in[27];
  const float* head_w   = (const float*)d_in[28];
  const float* head_b   = (const float*)d_in[29];
  (void)in_sizes; (void)n_in; (void)out_size; (void)ws_size;

  // workspace carve-up
  char* p = (char*)d_ws;
  auto carve = [&](size_t bytes) { void* r = (void*)p; p += (bytes + 255) & ~(size_t)255; return r; };
  const size_t n_inw  = (size_t)NL * 2 * DI * D_MODEL;   // 18,874,368
  const size_t n_outw = (size_t)NL * D_MODEL * DI;       //  9,437,184
  const size_t n_hw   = (size_t)NVOCAB * D_MODEL;        //    393,216
  unsigned short* wbf_in   = (unsigned short*)carve(n_inw  * 2);
  unsigned short* wbf_out  = (unsigned short*)carve(n_outw * 2);
  unsigned short* wbf_head = (unsigned short*)carve(n_hw   * 2);
  float*          condv    = (float*)         carve((size_t)NB * D_MODEL * 4);
  float*          x        = (float*)         carve((size_t)MROWS * D_MODEL * 4);
  unsigned short* hln      = (unsigned short*)carve((size_t)MROWS * D_MODEL * 2);
  unsigned short* xz       = (unsigned short*)carve((size_t)MROWS * 2 * DI * 2);
  unsigned short* xcs      = (unsigned short*)carve((size_t)MROWS * DI * 2);
  float*          dtb      = (float*)         carve((size_t)MROWS * 32 * 4);
  unsigned short* ybf      = (unsigned short*)carve((size_t)MROWS * DI * 2);

  // weight conversion (f32 -> bf16), once per call
  k_f32_to_bf16<<<(unsigned)((n_inw  + 255) / 256), 256, 0, stream>>>(in_w,  wbf_in,  n_inw);
  k_f32_to_bf16<<<(unsigned)((n_outw + 255) / 256), 256, 0, stream>>>(out_w, wbf_out, n_outw);
  k_f32_to_bf16<<<(unsigned)((n_hw   + 255) / 256), 256, 0, stream>>>(head_w, wbf_head, n_hw);

  // conditioning + embedding
  k_cond<<<(NB * D_MODEL + 255) / 256, 256, 0, stream>>>(
      emotion, style, keysig, modei, bpm, emo_emb, sty_emb, key_emb, mode_emb,
      bpm_w, bpm_b, cond_w, cond_b, condv);
  k_embed<<<(unsigned)(((size_t)MROWS * D_MODEL + 255) / 256), 256, 0, stream>>>(
      tokens, tok_emb, condv, pos_emb, x);

  const int lnBlocks = (MROWS * 32 + 255) / 256;   // one wave per row

  for (int i = 0; i < NL; ++i) {
    // LN(x) -> hln (bf16)
    k_layernorm<<<lnBlocks, 256, 0, stream>>>(x, ln_g + i * D_MODEL, ln_b + i * D_MODEL,
                                              hln, MROWS);
    // xz = hln @ in_w^T   [16384, 3072], bf16 out
    gemm_bf16_wmma<0><<<dim3((2 * DI) / 128, MROWS / 128), 256, 0, stream>>>(
        hln, wbf_in + (size_t)i * 2 * DI * D_MODEL, (void*)xz, nullptr, 2 * DI, D_MODEL);
    // depthwise conv + SiLU -> xcs (bf16)
    k_conv_silu<<<(unsigned)(((size_t)MROWS * DI + 255) / 256), 256, 0, stream>>>(
        xz, conv_w + (size_t)i * DI * 4, xcs);
    // dtb = xcs @ xproj_w^T   [16384, 32]
    k_xproj<<<MROWS / 8, 256, 0, stream>>>(xcs, xproj_w + (size_t)i * 32 * DI, dtb);
    // fused delta/bp + scan + D-residual + gating -> ybf (bf16)
    k_scan_gate<<<(NB * DI + 255) / 256, 256, 0, stream>>>(
        dtb, dt_w + (size_t)i * DI * DS, dt_b + (size_t)i * DI,
        bproj_w + (size_t)i * DI * DS, d_param + (size_t)i * DI, xcs, xz, ybf);
    // x += ybf @ out_w^T   [16384, 768], f32 accumulate
    gemm_bf16_wmma<1><<<dim3(D_MODEL / 128, MROWS / 128), 256, 0, stream>>>(
        ybf, wbf_out + (size_t)i * D_MODEL * DI, (void*)x, nullptr, D_MODEL, DI);
  }

  // final LN -> hln, then logits = hln @ head_w^T + head_b -> d_out (f32)
  k_layernorm<<<lnBlocks, 256, 0, stream>>>(x, final_g, final_b, hln, MROWS);
  gemm_bf16_wmma<2><<<dim3(NVOCAB / 128, MROWS / 128), 256, 0, stream>>>(
      hln, wbf_head, d_out, head_b, NVOCAB, D_MODEL);
}
